// SparseMoELayer_43748536877558
// MI455X (gfx1250) — compile-verified, matrix operand order checked by reading, not run on hardware
//
#include <hip/hip_runtime.h>
#include <hip/hip_bf16.h>
#include <math.h>

// ---------------------------------------------------------------------------
// Sparse MoE layer for MI455X (gfx1250, wave32, WMMA), routed dispatch.
//   x:[4,2048,1024] f32, Wg:[1024,8], bg:[8], W1:[8,1024,2048], b1:[8,2048],
//   W2:[8,2048,1024], b2:[8,1024]  ->  out:[4,2048,1024] f32 + entropy scalar
// ---------------------------------------------------------------------------

#define T_TOTAL 8192
#define DM 1024
#define DF 2048
#define NE 8
#define SCAT_SCALE 0.1f
#define TM 32                       // routed tokens per workgroup

typedef __attribute__((ext_vector_type(16))) __bf16 v16bf;
typedef __attribute__((ext_vector_type(8)))  __bf16 bf16x8;
typedef __attribute__((ext_vector_type(8)))  float  v8f;

#define W1_ELEMS ((size_t)NE * DM * DF)   // 16,777,216
#define W2_ELEMS ((size_t)NE * DF * DM)   // 16,777,216

// ---- fp32 -> bf16 round-to-nearest-even -----------------------------------
__device__ __forceinline__ __bf16 f2bf(float f) {
    union { float f; unsigned int u; } in;
    in.f = f;
    unsigned int u = in.u;
    unsigned int rounded = (u + 0x7FFFu + ((u >> 16) & 1u)) >> 16;
    union { unsigned short s; __bf16 b; } out;
    out.s = (unsigned short)rounded;
    return out.b;
}

// K index for element i (0..15) of a 16-bit A/B fragment, per ISA 7.12.2.
__device__ __forceinline__ int kmap(int i, int hf) {
    return (i < 8) ? (hf * 8 + i) : (16 + hf * 8 + (i - 8));
}

// A fragment (16x32 tile, row-major bf16 in LDS): two contiguous 16B runs.
__device__ __forceinline__ v16bf load_a_lds(const __bf16* __restrict__ s,
                                            int stride, int kBase, int hf, int m) {
    union { v16bf v; bf16x8 h[2]; } u;
    const __bf16* p = s + m * stride + kBase + hf * 8;
    u.h[0] = *(const bf16x8*)(p);
    u.h[1] = *(const bf16x8*)(p + 16);
    return u.v;
}

// Fallback B fragment (Kx16 tile) from global fp32 weights, on-the-fly bf16.
__device__ __forceinline__ v16bf load_b_gbl(const float* __restrict__ W,
                                            int rowStride, int kBase, int nBase,
                                            int hf, int n) {
    union { v16bf v; __bf16 e[16]; } u;
#pragma unroll
    for (int i = 0; i < 16; ++i)
        u.e[i] = f2bf(W[(kBase + kmap(i, hf)) * rowStride + nBase + n]);
    return u.v;
}

__device__ __forceinline__ v8f wmma_bf16(v16bf a, v16bf b, v8f c) {
    return __builtin_amdgcn_wmma_f32_16x16x32_bf16(false, a, false, b,
                                                   (short)0, c, false, false);
}

// ---------------------------------------------------------------------------
// Weight pre-pack: fp32 -> fragment-ordered bf16  [e][nTile][kTile][lane][16]
// ---------------------------------------------------------------------------
template <int KTILES, int NTILES, int ROWSTRIDE>
__global__ __launch_bounds__(256) void pack_weights_kernel(
    const float* __restrict__ W, __bf16* __restrict__ pk) {
    int idx = blockIdx.x * 256 + threadIdx.x;       // ((e*NTILES+nt)*KTILES+kt)*32+l
    const int l = idx & 31;
    int t = idx >> 5;
    const int kt = t % KTILES;
    t /= KTILES;
    const int nt = t % NTILES;
    const int e = t / NTILES;
    const int hf = l >> 4, n = l & 15;
    const float* src = W + (size_t)e * KTILES * 32 * ROWSTRIDE;
    __bf16* dst = pk + (size_t)idx * 16;
    union { v16bf v; __bf16 el[16]; } u;
#pragma unroll
    for (int i = 0; i < 16; ++i)
        u.el[i] = f2bf(src[(size_t)(kt * 32 + kmap(i, hf)) * ROWSTRIDE + nt * 16 + n]);
    *(v16bf*)dst = u.v;
}

// ---------------------------------------------------------------------------
// Kernel 1: per-token gating logits (x @ Wg + bg) and token L2 norm
// ---------------------------------------------------------------------------
__global__ __launch_bounds__(256) void gate_logits_kernel(
    const float* __restrict__ x, const float* __restrict__ Wg,
    const float* __restrict__ bg, float* __restrict__ logits,
    float* __restrict__ norms) {
    const int t = blockIdx.x;
    const int tid = threadIdx.x;
    float p[NE];
#pragma unroll
    for (int e = 0; e < NE; ++e) p[e] = 0.0f;
    float n2 = 0.0f;
    for (int d = tid; d < DM; d += 256) {
        float xv = x[(size_t)t * DM + d];
        n2 += xv * xv;
#pragma unroll
        for (int e = 0; e < NE; ++e) p[e] += xv * Wg[d * NE + e];
    }
    __shared__ float red[256];
#pragma unroll
    for (int e = 0; e < NE; ++e) {
        red[tid] = p[e];
        __syncthreads();
        for (int s = 128; s > 0; s >>= 1) {
            if (tid < s) red[tid] += red[tid + s];
            __syncthreads();
        }
        if (tid == 0) logits[t * NE + e] = red[0] + bg[e];
        __syncthreads();
    }
    red[tid] = n2;
    __syncthreads();
    for (int s = 128; s > 0; s >>= 1) {
        if (tid < s) red[tid] += red[tid + s];
        __syncthreads();
    }
    if (tid == 0) norms[t] = sqrtf(red[0]);
}

// ---------------------------------------------------------------------------
// Kernel 2: mean / std (ddof=1) of the 8192 token norms
// ---------------------------------------------------------------------------
__global__ __launch_bounds__(1024) void norm_stats_kernel(
    const float* __restrict__ norms, float* __restrict__ stats) {
    const int tid = threadIdx.x;
    float s = 0.0f, ss = 0.0f;
    for (int i = tid; i < T_TOTAL; i += 1024) {
        float v = norms[i];
        s += v;
        ss += v * v;
    }
    __shared__ float r1[1024], r2[1024];
    r1[tid] = s; r2[tid] = ss;
    __syncthreads();
    for (int k = 512; k > 0; k >>= 1) {
        if (tid < k) { r1[tid] += r1[tid + k]; r2[tid] += r2[tid + k]; }
        __syncthreads();
    }
    if (tid == 0) {
        float mean = r1[0] / (float)T_TOTAL;
        float var = (r2[0] - (float)T_TOTAL * mean * mean) / (float)(T_TOTAL - 1);
        stats[0] = mean;
        stats[1] = sqrtf(fmaxf(var, 0.0f));
    }
}

// ---------------------------------------------------------------------------
// Kernel 3: top-2 routing -> dense gate matrix [T, 8] + per-token entropy
// ---------------------------------------------------------------------------
__global__ __launch_bounds__(256) void routing_kernel(
    const float* __restrict__ logits, const float* __restrict__ norms,
    const float* __restrict__ stats, float* __restrict__ gates,
    float* __restrict__ ent) {
    const int t = blockIdx.x * blockDim.x + threadIdx.x;
    if (t >= T_TOTAL) return;
    const float mean = stats[0], sd = stats[1];
    const float adj = SCAT_SCALE * (norms[t] - mean) / (sd + 1e-6f);
    float l[NE];
#pragma unroll
    for (int e = 0; e < NE; ++e) l[e] = logits[t * NE + e] + adj;
    int i0 = 0; float v0 = l[0];
#pragma unroll
    for (int e = 1; e < NE; ++e)
        if (l[e] > v0) { v0 = l[e]; i0 = e; }
    int i1 = -1; float v1 = -3.4e38f;
#pragma unroll
    for (int e = 0; e < NE; ++e)
        if (e != i0 && l[e] > v1) { v1 = l[e]; i1 = e; }
    float e1 = __expf(v1 - v0);
    float inv = 1.0f / (1.0f + e1);
    float g0 = inv, g1 = e1 * inv;
    float g[NE];
#pragma unroll
    for (int e = 0; e < NE; ++e) g[e] = 0.0f;
    g[i0] += g0;
    g[i1] += g1;
#pragma unroll
    for (int e = 0; e < NE; ++e) gates[t * NE + e] = g[e];
    ent[t] = -(g0 * __logf(fmaxf(g0, 1e-8f)) + g1 * __logf(fmaxf(g1, 1e-8f)));
}

// ---------------------------------------------------------------------------
// Kernel 3b: deterministic entropy mean -> last element of d_out
// ---------------------------------------------------------------------------
__global__ __launch_bounds__(1024) void entropy_reduce_kernel(
    const float* __restrict__ ent, float* __restrict__ out_ent) {
    const int tid = threadIdx.x;
    float s = 0.0f;
    for (int i = tid; i < T_TOTAL; i += 1024) s += ent[i];
    __shared__ float red[1024];
    red[tid] = s;
    __syncthreads();
    for (int k = 512; k > 0; k >>= 1) {
        if (tid < k) red[tid] += red[tid + k];
        __syncthreads();
    }
    if (tid == 0) *out_ent = red[0] / (float)T_TOTAL;
}

// ---------------------------------------------------------------------------
// Kernel 3c: deterministic per-expert token compaction (block-wide scan).
// One block per expert; token order preserved -> pure function of gates.
// ---------------------------------------------------------------------------
__global__ __launch_bounds__(1024) void build_lists_kernel(
    const float* __restrict__ gates, int* __restrict__ tokList,
    int* __restrict__ cnt) {
    const int e = blockIdx.x;
    const int tid = threadIdx.x;
    __shared__ int pre[1024];
    __shared__ int base;
    if (tid == 0) base = 0;
    __syncthreads();
    for (int c0 = 0; c0 < T_TOTAL; c0 += 1024) {
        const int t = c0 + tid;
        const int flag = (gates[t * NE + e] != 0.0f) ? 1 : 0;
        pre[tid] = flag;
        __syncthreads();
        for (int off = 1; off < 1024; off <<= 1) {   // Hillis-Steele inclusive scan
            int v = (tid >= off) ? pre[tid - off] : 0;
            __syncthreads();
            pre[tid] += v;
            __syncthreads();
        }
        if (flag) tokList[e * T_TOTAL + base + pre[tid] - 1] = t;
        const int total = pre[1023];
        __syncthreads();
        if (tid == 0) base += total;
        __syncthreads();
    }
    if (tid == 0) cnt[e] = base;
}

// ---------------------------------------------------------------------------
// zero-init of the [T, DM] output block (d_ws/d_out are poisoned by harness)
// ---------------------------------------------------------------------------
__global__ __launch_bounds__(256) void zero_out_kernel(float4* __restrict__ out4) {
    out4[blockIdx.x * 256 + threadIdx.x] = make_float4(0.f, 0.f, 0.f, 0.f);
}

// ---------------------------------------------------------------------------
// Kernel 4: routed expert MLP. One block = 32 routed tokens of ONE expert.
// 8 waves; each B fragment feeds two M-tiles (2x FLOP per weight byte).
// LDS (dynamic 192 KB): x_bf16[32][1024] + H_bf16[32][2048].
// Output: exactly two atomic f32 adds per element onto zeroed out
// (commutative -> bit-deterministic).
// ---------------------------------------------------------------------------
template <bool PACKED>
__global__ __launch_bounds__(256) void moe_routed_kernel(
    const float* __restrict__ x, const float* __restrict__ W1,
    const float* __restrict__ b1, const float* __restrict__ W2,
    const float* __restrict__ b2, const __bf16* __restrict__ pk1,
    const __bf16* __restrict__ pk2, const float* __restrict__ gates,
    const int* __restrict__ tokList, const int* __restrict__ cnt,
    float* __restrict__ out) {
    const int e = blockIdx.y;
    const int n = cnt[e];
    const int row0 = blockIdx.x * TM;
    if (row0 >= n) return;                       // uniform early-exit

    extern __shared__ char smem[];
    __bf16* sX = (__bf16*)smem;                  // 32 x 1024 bf16 (64 KB)
    __bf16* sH = (__bf16*)(smem + TM * DM * 2);  // 32 x 2048 bf16 (128 KB)
    __shared__ int sTok[TM];
    __shared__ float sGate[TM];

    const int tid = threadIdx.x;
    const int wave = tid >> 5;
    const int lane = tid & 31;
    const int hf = lane >> 4;
    const int ln = lane & 15;

    if (tid < TM) {
        const int r = row0 + tid;
        const int t = (r < n) ? tokList[e * T_TOTAL + r] : 0;
        sTok[tid] = t;
        sGate[tid] = (r < n) ? gates[t * NE + e] : 0.0f;
    }
    __syncthreads();

    // gather routed x rows -> LDS bf16 (paired 32-bit stores)
    for (int i = tid; i < TM * DM / 2; i += 256) {
        const int r = i >> 9, c = (i & 511) * 2;
        const float2 xv = *(const float2*)&x[(size_t)sTok[r] * DM + c];
        union { unsigned int u; __bf16 b[2]; } o;
        o.b[0] = f2bf(xv.x);
        o.b[1] = f2bf(xv.y);
        *(unsigned int*)&sX[r * DM + c] = o.u;
    }
    __syncthreads();

    // ---- stage 1: H = relu(x @ W1_e + b1_e); wave owns F cols [wave*256,+256)
    const float* W1e = W1 + (size_t)e * DM * DF;
    for (int n2 = 0; n2 < 8; ++n2) {
        const int col0 = wave * 256 + n2 * 32;
        const int nt0 = col0 >> 4;
        v8f c00 = (v8f){}, c01 = (v8f){};        // M-tile 0, N-tiles 0/1
        v8f c10 = (v8f){}, c11 = (v8f){};        // M-tile 1, N-tiles 0/1
        const __bf16* fA = pk1 + ((((size_t)e * (DF / 16) + nt0) * (DM / 32)) * 32 + lane) * 16;
        const __bf16* fB = fA + (size_t)(DM / 32) * 32 * 16;   // nt0+1
#pragma unroll 2
        for (int kk = 0; kk < DM; kk += 32) {
            const v16bf a0 = load_a_lds(sX, DM, kk, hf, ln);
            const v16bf a1 = load_a_lds(sX, DM, kk, hf, 16 + ln);
            v16bf bA, bB;
            if constexpr (PACKED) {
                const int kt = kk >> 5;
                bA = *(const v16bf*)(fA + (size_t)kt * 512);
                bB = *(const v16bf*)(fB + (size_t)kt * 512);
                __builtin_prefetch(fA + (size_t)(kt + 1) * 512, 0, 1);
            } else {
                bA = load_b_gbl(W1e, DF, kk, col0, hf, ln);
                bB = load_b_gbl(W1e, DF, kk, col0 + 16, hf, ln);
            }
            c00 = wmma_bf16(a0, bA, c00);
            c01 = wmma_bf16(a0, bB, c01);
            c10 = wmma_bf16(a1, bA, c10);
            c11 = wmma_bf16(a1, bB, c11);
        }
        const float bias0 = b1[e * DF + col0 + ln];
        const float bias1 = b1[e * DF + col0 + 16 + ln];
#pragma unroll
        for (int r = 0; r < 8; ++r) {
            const int m = r + hf * 8;
            sH[m * DF + col0 + ln]             = f2bf(fmaxf(c00[r] + bias0, 0.0f));
            sH[m * DF + col0 + 16 + ln]        = f2bf(fmaxf(c01[r] + bias1, 0.0f));
            sH[(16 + m) * DF + col0 + ln]      = f2bf(fmaxf(c10[r] + bias0, 0.0f));
            sH[(16 + m) * DF + col0 + 16 + ln] = f2bf(fmaxf(c11[r] + bias1, 0.0f));
        }
    }
    __syncthreads();   // H complete before any wave consumes it

    // ---- stage 2: y = H @ W2_e + b2_e ; out += gate * y (atomic, 2/elem total)
    for (int m2 = 0; m2 < 4; ++m2) {
        const int col0 = wave * 128 + m2 * 32;
        const int nt0 = col0 >> 4;
        v8f c00 = (v8f){}, c01 = (v8f){};
        v8f c10 = (v8f){}, c11 = (v8f){};
        const __bf16* fA = pk2 + ((((size_t)e * (DM / 16) + nt0) * (DF / 32)) * 32 + lane) * 16;
        const __bf16* fB = fA + (size_t)(DF / 32) * 32 * 16;   // nt0+1
        const float* W2e = W2 + (size_t)e * DF * DM;
#pragma unroll 2
        for (int kk = 0; kk < DF; kk += 32) {
            const v16bf a0 = load_a_lds(sH, DF, kk, hf, ln);
            const v16bf a1 = load_a_lds(sH, DF, kk, hf, 16 + ln);
            v16bf bA, bB;
            if constexpr (PACKED) {
                const int kt = kk >> 5;
                bA = *(const v16bf*)(fA + (size_t)kt * 512);
                bB = *(const v16bf*)(fB + (size_t)kt * 512);
                __builtin_prefetch(fA + (size_t)(kt + 1) * 512, 0, 1);
            } else {
                bA = load_b_gbl(W2e, DM, kk, col0, hf, ln);
                bB = load_b_gbl(W2e, DM, kk, col0 + 16, hf, ln);
            }
            c00 = wmma_bf16(a0, bA, c00);
            c01 = wmma_bf16(a0, bB, c01);
            c10 = wmma_bf16(a1, bA, c10);
            c11 = wmma_bf16(a1, bB, c11);
        }
        const float bias0 = b2[e * DM + col0 + ln];
        const float bias1 = b2[e * DM + col0 + 16 + ln];
#pragma unroll
        for (int mt = 0; mt < 2; ++mt) {
            const v8f* cA = mt ? &c10 : &c00;
            const v8f* cB = mt ? &c11 : &c01;
#pragma unroll
            for (int r = 0; r < 8; ++r) {
                const int row = mt * 16 + r + hf * 8;
                if (row0 + row < n) {
                    const float g = sGate[row];
                    float* op = out + (size_t)sTok[row] * DM + col0;
                    atomicAdd(op + ln,      g * ((*cA)[r] + bias0));
                    atomicAdd(op + 16 + ln, g * ((*cB)[r] + bias1));
                }
            }
        }
    }
}

// ---------------------------------------------------------------------------
// host entry
// ---------------------------------------------------------------------------
extern "C" void kernel_launch(void* const* d_in, const int* in_sizes, int n_in,
                              void* d_out, int out_size, void* d_ws, size_t ws_size,
                              hipStream_t stream) {
    (void)in_sizes; (void)n_in;
    const float* x  = (const float*)d_in[0];
    const float* Wg = (const float*)d_in[1];
    const float* bg = (const float*)d_in[2];
    const float* W1 = (const float*)d_in[3];
    const float* b1 = (const float*)d_in[4];
    const float* W2 = (const float*)d_in[5];
    const float* b2 = (const float*)d_in[6];
    float* out = (float*)d_out;

    float* ws     = (float*)d_ws;
    float* logits = ws;                 // 8192*8 f32
    float* norms  = ws + 65536;         // 8192 f32
    float* stats  = ws + 73728;         // 2 f32 (+pad)
    float* gates  = ws + 73736;         // 8192*8 f32
    float* ent    = ws + 139272;        // 8192 f32  (ends at float 147464)
    int* tokList  = (int*)((char*)d_ws + 589856);   // 8*8192 int
    int* cnt      = (int*)((char*)d_ws + 852000);   // 8 int

    const size_t pk1Off = 1u << 20;                         // 1 MB
    const size_t pkBytes = W1_ELEMS * 2;                    // 32 MB each
    __bf16* pk1 = (__bf16*)((char*)d_ws + pk1Off);
    __bf16* pk2 = pk1 + W1_ELEMS;
    const bool packed = ws_size >= pk1Off + 2 * pkBytes;    // deterministic branch

    // gating pipeline
    gate_logits_kernel<<<T_TOTAL, 256, 0, stream>>>(x, Wg, bg, logits, norms);
    norm_stats_kernel<<<1, 1024, 0, stream>>>(norms, stats);
    routing_kernel<<<T_TOTAL / 256, 256, 0, stream>>>(logits, norms, stats, gates, ent);
    entropy_reduce_kernel<<<1, 1024, 0, stream>>>(ent, out + (out_size - 1));
    build_lists_kernel<<<NE, 1024, 0, stream>>>(gates, tokList, cnt);
    zero_out_kernel<<<(T_TOTAL * DM / 4) / 256, 256, 0, stream>>>((float4*)out);

    const int dynLds = (TM * DM + TM * DF) * 2;   // 192 KB (<= 320 KB/WG on CDNA5)
    dim3 grid(T_TOTAL / TM, NE);                  // worst-case tiles per expert
    if (packed) {
        pack_weights_kernel<DM / 32, DF / 16, DF>
            <<<(int)(W1_ELEMS / 16 / 256), 256, 0, stream>>>(W1, pk1);
        pack_weights_kernel<DF / 32, DM / 16, DM>
            <<<(int)(W2_ELEMS / 16 / 256), 256, 0, stream>>>(W2, pk2);
        hipFuncSetAttribute((const void*)moe_routed_kernel<true>,
                            hipFuncAttributeMaxDynamicSharedMemorySize, dynLds);
        moe_routed_kernel<true><<<grid, 256, dynLds, stream>>>(
            x, W1, b1, W2, b2, pk1, pk2, gates, tokList, cnt, out);
    } else {
        hipFuncSetAttribute((const void*)moe_routed_kernel<false>,
                            hipFuncAttributeMaxDynamicSharedMemorySize, dynLds);
        moe_routed_kernel<false><<<grid, 256, dynLds, stream>>>(
            x, W1, b1, W2, b2, pk1, pk2, gates, tokList, cnt, out);
    }
}
